// GlobalAttentionModule_74491912782023
// MI455X (gfx1250) — matrix operand order repaired and checked by reading, not compiled, not run on hardware
//
#include <hip/hip_runtime.h>
#include <stdint.h>

// ---------- types ----------
typedef __bf16   v16bf __attribute__((ext_vector_type(16)));
typedef float    v8f   __attribute__((ext_vector_type(8)));
typedef uint32_t u32x4 __attribute__((ext_vector_type(4)));

union Frag { u32x4 u[2]; v16bf v; };   // 32 bytes = 16 bf16 = one WMMA A/B fragment per lane

__device__ __forceinline__ v8f wmma_bf16(v16bf a, v16bf b, v8f c) {
  // D = A(16x32 bf16) * B(32x16 bf16) + C(16x16 f32)
  return __builtin_amdgcn_wmma_f32_16x16x32_bf16(false, a, false, b, (short)0, c, false, false);
}

#define H   512
#define C_  32
#define NH  8
#define HD  64

// ============================================================================
// 1) prep: transpose weights to bf16 Wt[n][k], transpose centroids, norms, zero sums
// ============================================================================
__global__ __launch_bounds__(256) void prep_kernel(
    const float* __restrict__ Wq, const float* __restrict__ Wk,
    const float* __restrict__ Wv, const float* __restrict__ Wo,
    const float* __restrict__ cent,
    __bf16* __restrict__ Wtq, __bf16* __restrict__ Wtk,
    __bf16* __restrict__ Wtv, __bf16* __restrict__ Wto,
    float* __restrict__ ct, float* __restrict__ cnorm,
    float* __restrict__ sums, float* __restrict__ counts) {
  int tid = blockIdx.x * blockDim.x + threadIdx.x;
  int nt  = gridDim.x * blockDim.x;
  for (int i = tid; i < H * H; i += nt) {
    int n = i >> 9, k = i & (H - 1);
    Wtq[i] = (__bf16)Wq[k * H + n];
    Wtk[i] = (__bf16)Wk[k * H + n];
    Wtv[i] = (__bf16)Wv[k * H + n];
    Wto[i] = (__bf16)Wo[k * H + n];
  }
  for (int i = tid; i < H * C_; i += nt) {   // ct[d*32+c] = cent[c][d]
    int d = i >> 5, c = i & 31;
    ct[i] = cent[c * H + d];
  }
  for (int i = tid; i < C_ * H; i += nt) sums[i] = 0.f;
  if (tid < C_) {
    counts[tid] = 0.f;
    float s = 0.f;
    for (int d = 0; d < H; ++d) { float v = cent[tid * H + d]; s += v * v; }
    cnorm[tid] = s;
  }
}

// ============================================================================
// 2) assign: wave per node, lane = centroid; LDS partial segment sums per WG
// ============================================================================
__global__ __launch_bounds__(256) void assign_kernel(
    const float* __restrict__ x, const float* __restrict__ ct,
    const float* __restrict__ cnorm,
    float* __restrict__ sums, float* __restrict__ counts) {
  extern __shared__ float lmem[];          // 32*512 sums + 32 counts
  float* lsum = lmem;
  float* lcnt = lmem + C_ * H;
  int tid = threadIdx.x, lane = tid & 31, wave = tid >> 5;
  for (int i = tid; i < C_ * H; i += 256) lsum[i] = 0.f;
  if (tid < C_) lcnt[tid] = 0.f;
  __syncthreads();

  int base = blockIdx.x * 1024 + wave * 128;   // 128 nodes per wave
  for (int n = 0; n < 128; ++n) {
    const float* xr = x + (size_t)(base + n) * H;
    float xreg[16];
    float acc = 0.f;                            // dot(x, centroid[lane])
#pragma unroll
    for (int ch = 0; ch < 16; ++ch) {
      float xv = xr[ch * 32 + lane];
      xreg[ch] = xv;
#pragma unroll
      for (int j = 0; j < 32; ++j) {
        float s = __shfl(xv, j);
        acc = fmaf(s, ct[(ch * 32 + j) * 32 + lane], acc);
      }
    }
    float d2 = cnorm[lane] - 2.f * acc;         // ||x||^2 constant per row: drop
    int best = lane;
#pragma unroll
    for (int m = 16; m >= 1; m >>= 1) {         // argmin, first-index tiebreak
      float od = __shfl_xor(d2, m);
      int   ob = __shfl_xor(best, m);
      if (od < d2 || (od == d2 && ob < best)) { d2 = od; best = ob; }
    }
    if (lane == 0) atomicAdd(&lcnt[best], 1.f);
#pragma unroll
    for (int ch = 0; ch < 16; ++ch)
      atomicAdd(&lsum[best * H + ch * 32 + lane], xreg[ch]);
  }
  __syncthreads();
  for (int i = tid; i < C_ * H; i += 256) atomicAdd(&sums[i], lsum[i]);
  if (tid < C_) atomicAdd(&counts[tid], lcnt[tid]);
}

// ============================================================================
// 3) EMA centroid update -> bf16
// ============================================================================
__global__ __launch_bounds__(256) void update_kernel(
    const float* __restrict__ cent, const float* __restrict__ sums,
    const float* __restrict__ counts, __bf16* __restrict__ cnew) {
  int i = blockIdx.x * 256 + threadIdx.x;
  if (i >= C_ * H) return;
  int c = i >> 9;
  float cnt  = counts[c];
  float mean = sums[i] / fmaxf(cnt, 1.f);
  float old  = cent[i];
  cnew[i] = (__bf16)(cnt > 0.f ? (0.99f * old + 0.01f * mean) : old);
}

// ============================================================================
// 4) k = cnew@Wk + bk (scaled by 1/8), v = cnew@Wv + bv; k row-major, v transposed
// ============================================================================
__global__ __launch_bounds__(256) void kv_kernel(
    const __bf16* __restrict__ cnew,
    const __bf16* __restrict__ Wtk, const __bf16* __restrict__ Wtv,
    const float* __restrict__ bk, const float* __restrict__ bv,
    __bf16* __restrict__ k_bf, __bf16* __restrict__ vT_bf) {
  int lane = threadIdx.x & 31, wave = threadIdx.x >> 5;
  int wg = blockIdx.x * 8 + wave;      // 64 waves: mt in [0,2), nt in [0,32)
  int mt = wg >> 5, nt = wg & 31;
  int lo = lane & 15, hi = lane >> 4;
  v8f acck = {}, accv = {};
  for (int kb = 0; kb < H; kb += 32) {
    Frag a, b1, b2;
    const __bf16* ap = cnew + (mt * 16 + lo) * H + kb + hi * 16;
    a.u[0] = *(const u32x4*)ap;  a.u[1] = *(const u32x4*)(ap + 8);
    const __bf16* bp = Wtk + (nt * 16 + lo) * H + kb + hi * 16;
    b1.u[0] = *(const u32x4*)bp; b1.u[1] = *(const u32x4*)(bp + 8);
    const __bf16* cp = Wtv + (nt * 16 + lo) * H + kb + hi * 16;
    b2.u[0] = *(const u32x4*)cp; b2.u[1] = *(const u32x4*)(cp + 8);
    acck = wmma_bf16(a.v, b1.v, acck);
    accv = wmma_bf16(a.v, b2.v, accv);
  }
  int ncol = nt * 16 + lo;
  float bkv = bk[ncol] * 0.125f, bvv = bv[ncol];
#pragma unroll
  for (int r = 0; r < 8; ++r) {
    int row = mt * 16 + (hi ? 8 + r : r);
    k_bf[row * H + ncol]   = (__bf16)(acck[r] * 0.125f + bkv);  // fold 1/sqrt(HD)
    vT_bf[ncol * C_ + row] = (__bf16)(accv[r] + bvv);
  }
}

// ============================================================================
// 5) fused: q-GEMM -> attention (scores/softmax/ctx) -> out-GEMM + residual
//    64 nodes per WG, 8 waves. LDS: xb/ctx 64KB | qb 64KB | attn scratch 8KB
// ============================================================================
__global__ __launch_bounds__(256) void fused_main(
    const float* __restrict__ x,
    const __bf16* __restrict__ Wtq, const float* __restrict__ bq,
    const __bf16* __restrict__ k_bf, const __bf16* __restrict__ vT_bf,
    const __bf16* __restrict__ Wto, const float* __restrict__ bo,
    float* __restrict__ y) {
  extern __shared__ char smem[];
  __bf16* xb   = (__bf16*)smem;                  // 64x512 bf16: x, later ctx
  __bf16* qb   = (__bf16*)(smem + 65536);        // 64x512 bf16: q
  __bf16* attn = (__bf16*)(smem + 131072);       // 8 waves x 16x32 bf16
  int tid = threadIdx.x, lane = tid & 31, wave = tid >> 5;
  int lo = lane & 15, hi = lane >> 4;
  size_t rowbase = (size_t)blockIdx.x * 64;

  // stage x tile -> bf16 LDS
  for (int i = tid; i < 64 * H; i += 256)
    xb[i] = (__bf16)x[rowbase * H + i];
  __syncthreads();

  // ---- q = xb @ Wtq + bq (wave: mt = w&3, 16 n-tiles) ----
  {
    int mt = wave & 3, nt0 = (wave >> 2) * 16;
    for (int g = 0; g < 4; ++g) {
      v8f acc[4] = {};
      for (int kb = 0; kb < H; kb += 32) {
        Frag a;
        const __bf16* ap = xb + (mt * 16 + lo) * H + kb + hi * 16;
        a.u[0] = *(const u32x4*)ap; a.u[1] = *(const u32x4*)(ap + 8);
#pragma unroll
        for (int t = 0; t < 4; ++t) {
          Frag b;
          const __bf16* bp = Wtq + ((nt0 + g * 4 + t) * 16 + lo) * H + kb + hi * 16;
          b.u[0] = *(const u32x4*)bp; b.u[1] = *(const u32x4*)(bp + 8);
          acc[t] = wmma_bf16(a.v, b.v, acc[t]);
        }
      }
#pragma unroll
      for (int t = 0; t < 4; ++t) {
        int ncol = (nt0 + g * 4 + t) * 16 + lo;
        float bias = bq[ncol];
#pragma unroll
        for (int r = 0; r < 8; ++r) {
          int row = mt * 16 + (hi ? 8 + r : r);
          qb[row * H + ncol] = (__bf16)(acc[t][r] + bias);
        }
      }
    }
  }
  __syncthreads();

  // ---- attention: 32 units (8 heads x 4 m-tiles), 4 per wave ----
  __bf16* asc = attn + wave * 512;
  for (int u = wave; u < 32; u += 8) {
    int h = u >> 2, mt = u & 3;
    // scores[16x32] = q_h[16x64] @ k_h^T  (k already scaled by 1/8)
    v8f s0 = {}, s1 = {};
#pragma unroll
    for (int kb2 = 0; kb2 < HD; kb2 += 32) {
      Frag a, b0, b1;
      const __bf16* ap = qb + (mt * 16 + lo) * H + h * HD + kb2 + hi * 16;
      a.u[0] = *(const u32x4*)ap; a.u[1] = *(const u32x4*)(ap + 8);
      const __bf16* p0 = k_bf + lo * H        + h * HD + kb2 + hi * 16;
      const __bf16* p1 = k_bf + (16 + lo) * H + h * HD + kb2 + hi * 16;
      b0.u[0] = *(const u32x4*)p0; b0.u[1] = *(const u32x4*)(p0 + 8);
      b1.u[0] = *(const u32x4*)p1; b1.u[1] = *(const u32x4*)(p1 + 8);
      s0 = wmma_bf16(a.v, b0.v, s0);
      s1 = wmma_bf16(a.v, b1.v, s1);
    }
    // softmax over 32 cols: rows live in 16-lane halves -> xor masks 8..1
#pragma unroll
    for (int r = 0; r < 8; ++r) {
      float v0 = s0[r], v1 = s1[r];
      float m = fmaxf(v0, v1);
#pragma unroll
      for (int msk = 8; msk >= 1; msk >>= 1) m = fmaxf(m, __shfl_xor(m, msk));
      float e0 = __expf(v0 - m), e1 = __expf(v1 - m);
      float s = e0 + e1;
#pragma unroll
      for (int msk = 8; msk >= 1; msk >>= 1) s += __shfl_xor(s, msk);
      float inv = 1.f / s;
      int row = (hi ? 8 + r : r);
      asc[row * 32 + lo]      = (__bf16)(e0 * inv);
      asc[row * 32 + 16 + lo] = (__bf16)(e1 * inv);
    }
    // ctx[16x64] = attn[16x32] @ v_h[32x64]; store bf16 into xb (x retired)
    Frag a;
    const __bf16* ap = asc + lo * 32 + hi * 16;
    a.u[0] = *(const u32x4*)ap; a.u[1] = *(const u32x4*)(ap + 8);
#pragma unroll
    for (int nt = 0; nt < 4; ++nt) {
      Frag b;
      const __bf16* bp = vT_bf + (h * HD + nt * 16 + lo) * C_ + hi * 16;
      b.u[0] = *(const u32x4*)bp; b.u[1] = *(const u32x4*)(bp + 8);
      v8f cz = {};
      v8f c = wmma_bf16(a.v, b.v, cz);
#pragma unroll
      for (int r = 0; r < 8; ++r) {
        int row = mt * 16 + (hi ? 8 + r : r);
        xb[row * H + h * HD + nt * 16 + lo] = (__bf16)c[r];
      }
    }
  }
  __syncthreads();

  // ---- out = ctx @ Wto + bo + residual -> y (pre-LN) ----
  {
    int mt = wave & 3, nt0 = (wave >> 2) * 16;
    for (int g = 0; g < 4; ++g) {
      v8f acc[4] = {};
      for (int kb = 0; kb < H; kb += 32) {
        Frag a;
        const __bf16* ap = xb + (mt * 16 + lo) * H + kb + hi * 16;
        a.u[0] = *(const u32x4*)ap; a.u[1] = *(const u32x4*)(ap + 8);
#pragma unroll
        for (int t = 0; t < 4; ++t) {
          Frag b;
          const __bf16* bp = Wto + ((nt0 + g * 4 + t) * 16 + lo) * H + kb + hi * 16;
          b.u[0] = *(const u32x4*)bp; b.u[1] = *(const u32x4*)(bp + 8);
          acc[t] = wmma_bf16(a.v, b.v, acc[t]);
        }
      }
#pragma unroll
      for (int t = 0; t < 4; ++t) {
        int ncol = (nt0 + g * 4 + t) * 16 + lo;
        float bias = bo[ncol];
#pragma unroll
        for (int r = 0; r < 8; ++r) {
          int row = mt * 16 + (hi ? 8 + r : r);
          size_t gi = (rowbase + row) * H + ncol;
          y[gi] = acc[t][r] + bias + x[gi];
        }
      }
    }
  }
}

// ============================================================================
// 6) LayerNorm in place on y (one wave per row)
// ============================================================================
__global__ __launch_bounds__(256) void ln_kernel(
    float* __restrict__ y, const float* __restrict__ lnw,
    const float* __restrict__ lnb) {
  int lane = threadIdx.x & 31, wave = threadIdx.x >> 5;
  size_t row = (size_t)blockIdx.x * 8 + wave;
  float* yr = y + row * H;
  float v[16];
  float s = 0.f;
#pragma unroll
  for (int j = 0; j < 16; ++j) { v[j] = yr[j * 32 + lane]; s += v[j]; }
#pragma unroll
  for (int m = 16; m >= 1; m >>= 1) s += __shfl_xor(s, m);
  float mu = s * (1.f / 512.f);
  float var = 0.f;
#pragma unroll
  for (int j = 0; j < 16; ++j) { float d = v[j] - mu; var += d * d; }
#pragma unroll
  for (int m = 16; m >= 1; m >>= 1) var += __shfl_xor(var, m);
  float inv = rsqrtf(var * (1.f / 512.f) + 1e-5f);
#pragma unroll
  for (int j = 0; j < 16; ++j) {
    int cc = j * 32 + lane;
    yr[cc] = (v[j] - mu) * inv * lnw[cc] + lnb[cc];
  }
}

// ============================================================================
extern "C" void kernel_launch(void* const* d_in, const int* in_sizes, int n_in,
                              void* d_out, int out_size, void* d_ws, size_t ws_size,
                              hipStream_t stream) {
  const float* x    = (const float*)d_in[0];
  const float* cent = (const float*)d_in[1];
  const float* Wq   = (const float*)d_in[2];
  const float* bq   = (const float*)d_in[3];
  const float* Wk   = (const float*)d_in[4];
  const float* bk   = (const float*)d_in[5];
  const float* Wv   = (const float*)d_in[6];
  const float* bv   = (const float*)d_in[7];
  const float* Wo   = (const float*)d_in[8];
  const float* bo   = (const float*)d_in[9];
  const float* lnw  = (const float*)d_in[10];
  const float* lnb  = (const float*)d_in[11];
  float* out = (float*)d_out;
  int N = in_sizes[0] / H;   // 262144

  // workspace carve (~2.3 MB)
  char*  w   = (char*)d_ws;
  size_t off = 0;
  auto alloc = [&](size_t bytes) { void* p = w + off; off += (bytes + 255) & ~(size_t)255; return p; };
  __bf16* Wtq   = (__bf16*)alloc(H * H * 2);
  __bf16* Wtk   = (__bf16*)alloc(H * H * 2);
  __bf16* Wtv   = (__bf16*)alloc(H * H * 2);
  __bf16* Wto   = (__bf16*)alloc(H * H * 2);
  float*  ct    = (float*) alloc(H * C_ * 4);
  float*  cnorm = (float*) alloc(C_ * 4);
  float*  sums  = (float*) alloc(C_ * H * 4);
  float*  cnts  = (float*) alloc(C_ * 4);
  __bf16* cnew  = (__bf16*)alloc(C_ * H * 2);
  __bf16* k_bf  = (__bf16*)alloc(C_ * H * 2);
  __bf16* vT_bf = (__bf16*)alloc(H * C_ * 2);

  prep_kernel<<<256, 256, 0, stream>>>(Wq, Wk, Wv, Wo, cent,
                                       Wtq, Wtk, Wtv, Wto, ct, cnorm, sums, cnts);
  assign_kernel<<<N / 1024, 256, (C_ * H + C_) * 4, stream>>>(x, ct, cnorm, sums, cnts);
  update_kernel<<<(C_ * H + 255) / 256, 256, 0, stream>>>(cent, sums, cnts, cnew);
  kv_kernel<<<8, 256, 0, stream>>>(cnew, Wtk, Wtv, bk, bv, k_bf, vT_bf);
  fused_main<<<N / 64, 256, 65536 + 65536 + 8192, stream>>>(
      x, Wtq, bq, k_bf, vT_bf, Wto, bo, out);
  ln_kernel<<<N / 8, 256, 0, stream>>>(out, lnw, lnb);
}
